// TOSCA_45578192945199
// MI455X (gfx1250) — compile-verified, matrix operand order checked by reading, not run on hardware
//
#include <hip/hip_runtime.h>

// ---------------------------------------------------------------------------
// CDNA5 (gfx1250) WMMA helpers: v_wmma_f32_16x16x32_f16, wave32, 16x16 tiles.
// Tiles are stored in "fragment order": for lane l (= hh*16 + m) and K-block
// kt, the 16 f16 fragment elements are contiguous at (kt*32 + l)*16, so a
// fragment load is one aligned 32-byte vector load (2x ds_load_b128).
// Element (m, k) maps to offset frag_off(m,k) per the ISA 16-bit A layout:
//   kk = k%32 ; hh = (kk>>3)&1 ; i = (kk&7) | ((kk>>4)<<3)
// C/D fragment: c[j] = D[j + 8*(lane>>4)][lane&15].
// ---------------------------------------------------------------------------
typedef __attribute__((ext_vector_type(16))) _Float16 v16h;
typedef __attribute__((ext_vector_type(8)))  float    v8f;

#define DEVINL __device__ __forceinline__

DEVINL int frag_off(int m, int k) {
  const int kt = k >> 5, kk = k & 31;
  const int hh = (kk >> 3) & 1;
  const int i  = (kk & 7) | ((kk >> 4) << 3);
  return ((kt << 5) | (hh << 4) | m) * 16 + i;
}

DEVINL v16h load_frag(const _Float16* tile, int lane, int kt) {
  return *(const v16h*)(tile + ((kt << 5) + lane) * 16);
}

// weights packed as blocks bi = nt*ktot + kt, each block 32 lanes x 16 elems
DEVINL v16h load_fragW(const _Float16* w, int lane, int kt, int nt, int ktot) {
  return *(const v16h*)(w + (((nt * ktot + kt) << 5) + lane) * 16);
}

DEVINL v8f wmma16(v16h a, v16h b, v8f c) {
  return __builtin_amdgcn_wmma_f32_16x16x32_f16(false, a, false, b, (short)0, c,
                                                false, false);
}

// fast silu: one v_exp_f32 + one v_rcp_f32 (no IEEE divide expansion)
DEVINL float siluf(float x) {
  return x * __builtin_amdgcn_rcpf(1.f + __expf(-x));
}

// ---------------------------------------------------------------------------
// fp32 -> f16 weight repack into fragment order: src [K][Nn] row-major ->
// dst fragment-ordered [KP/32 x NP/16 blocks], zero padded.
// ---------------------------------------------------------------------------
__global__ void k_cvt(const float* __restrict__ src, _Float16* __restrict__ dst,
                      int K, int Nn, int KP, int NP) {
  int d = blockIdx.x * blockDim.x + threadIdx.x;
  if (d >= KP * NP) return;
  const int i    = d & 15;
  const int lane = (d >> 4) & 31;
  const int blk  = d >> 9;
  const int ktot = KP >> 5;
  const int kt = blk % ktot, nt = blk / ktot;
  const int hh = lane >> 4, nn = lane & 15;
  const int kk = (i & 7) + (hh << 3) + ((i >> 3) << 4);
  const int k = (kt << 5) + kk, n = (nt << 4) + nn;
  dst[d] = (_Float16)((k < K && n < Nn) ? src[k * Nn + n] : 0.f);
}

// ---------------------------------------------------------------------------
// Normalization / feature prep
// ---------------------------------------------------------------------------
__global__ void k_centroid(const float* __restrict__ pos, const int* __restrict__ batch,
                           float* csum, float* ccnt, int N) {
  int i = blockIdx.x * blockDim.x + threadIdx.x;
  if (i >= N) return;
  int b = batch[i];
  atomicAdd(&csum[b * 3 + 0], pos[i * 3 + 0]);
  atomicAdd(&csum[b * 3 + 1], pos[i * 3 + 1]);
  atomicAdd(&csum[b * 3 + 2], pos[i * 3 + 2]);
  atomicAdd(&ccnt[b], 1.f);
}

__global__ void k_norm1(const float* __restrict__ pos, const int* __restrict__ batch,
                        const float* __restrict__ csum, const float* __restrict__ ccnt,
                        float* __restrict__ coord, unsigned int* mrad, int N) {
  int i = blockIdx.x * blockDim.x + threadIdx.x;
  if (i >= N) return;
  int b = batch[i];
  float inv = 1.f / fmaxf(ccnt[b], 1.f);
  float x = pos[i * 3 + 0] - csum[b * 3 + 0] * inv;
  float y = pos[i * 3 + 1] - csum[b * 3 + 1] * inv;
  float z = pos[i * 3 + 2] - csum[b * 3 + 2] * inv;
  coord[i * 3 + 0] = x; coord[i * 3 + 1] = y; coord[i * 3 + 2] = z;
  float r = sqrtf(x * x + y * y + z * z);
  atomicMax(&mrad[b], __float_as_uint(r));  // r >= 0: uint order == float order
}

__global__ void k_norm2(float* __restrict__ coord, const int* __restrict__ batch,
                        const unsigned int* __restrict__ mrad, int N) {
  int i = blockIdx.x * blockDim.x + threadIdx.x;
  if (i >= N) return;
  float inv = 1.f / __uint_as_float(mrad[batch[i]]);
  coord[i * 3 + 0] *= inv; coord[i * 3 + 1] *= inv; coord[i * 3 + 2] *= inv;
}

__global__ void k_farea(const float* __restrict__ coord, const int* __restrict__ face,
                        float* __restrict__ farea, int F) {
  int f = blockIdx.x * blockDim.x + threadIdx.x;
  if (f >= F) return;
  int a = face[f], b = face[F + f], c = face[2 * F + f];
  float ax = coord[a * 3], ay = coord[a * 3 + 1], az = coord[a * 3 + 2];
  float v1x = coord[b * 3] - ax, v1y = coord[b * 3 + 1] - ay, v1z = coord[b * 3 + 2] - az;
  float v2x = coord[c * 3] - ax, v2y = coord[c * 3 + 1] - ay, v2z = coord[c * 3 + 2] - az;
  float cx = v1y * v2z - v1z * v2y;
  float cy = v1z * v2x - v1x * v2z;
  float cz = v1x * v2y - v1y * v2x;
  farea[f] = 0.5f * sqrtf(cx * cx + cy * cy + cz * cz);
}

__global__ void k_v2f(const int* __restrict__ v2fv, const int* __restrict__ v2ff,
                      const float* __restrict__ farea, float* asum, float* acnt, int V) {
  int i = blockIdx.x * blockDim.x + threadIdx.x;
  if (i >= V) return;
  int v = v2fv[i];
  atomicAdd(&asum[v], farea[v2ff[i]]);
  atomicAdd(&acnt[v], 1.f);
}

__global__ void k_feat(const float* __restrict__ asum, const float* __restrict__ acnt,
                       const float* __restrict__ fw, const float* __restrict__ fb,
                       float* __restrict__ h, int N) {
  int i = blockIdx.x * blockDim.x + threadIdx.x;
  if (i >= N) return;
  float area = asum[i] / fmaxf(acnt[i], 1.f);
#pragma unroll
  for (int f = 0; f < 16; ++f) h[i * 16 + f] = area * fw[f] + fb[f];
}

__global__ void k_cntrow(const int* __restrict__ ei, float* cnt, int E) {
  int e = blockIdx.x * blockDim.x + threadIdx.x;
  if (e >= E) return;
  atomicAdd(&cnt[ei[e]], 1.f);
}

__global__ void k_coordupd(const float* __restrict__ cin, const float* __restrict__ delta,
                           const float* __restrict__ cnt, float* __restrict__ cout, int N) {
  int i = blockIdx.x * blockDim.x + threadIdx.x;
  if (i >= N) return;
  float inv = 1.f / fmaxf(cnt[i], 1.f);
#pragma unroll
  for (int d = 0; d < 3; ++d) cout[i * 3 + d] = cin[i * 3 + d] + delta[i * 3 + d] * inv;
}

// ---------------------------------------------------------------------------
// Fused EGCL edge kernel: one wave per 16-edge tile.
//   ef = [h[row], h[col], radial, edge_attr]   (f16 tile in LDS, frag order)
//   m  = silu(silu(ef@We1+be1)@We2+be2)         -> agg atomics + LDS
//   ct = (silu(m@Wc1+bc1)) . Wc2                -> coord-delta atomics
// All GEMMs are v_wmma_f32_16x16x32_f16, fp32 accumulate.
// ---------------------------------------------------------------------------
template <int NIN, int NH>
__global__ __launch_bounds__(32) void k_edge(
    const int* __restrict__ ei, const float* __restrict__ eattr,
    const float* __restrict__ h, const float* __restrict__ coord,
    const _Float16* __restrict__ We1, const float* __restrict__ be1,
    const _Float16* __restrict__ We2, const float* __restrict__ be2,
    const _Float16* __restrict__ Wc1, const float* __restrict__ bc1,
    const float* __restrict__ Wc2,
    float* __restrict__ agg, float* __restrict__ cdelta, int E) {
  constexpr int K1  = 2 * NIN + 2;
  constexpr int K1P = (K1 + 31) & ~31;
  constexpr int NHK = (NH < 32) ? 32 : NH;

  __shared__ alignas(32) _Float16 s_ef[16 * K1P];
  __shared__ alignas(32) _Float16 s_m1[16 * NHK];
  __shared__ alignas(32) _Float16 s_m2[16 * NHK];
  __shared__ float s_t[16][NH];
  __shared__ float s_diff[16][3];
  __shared__ int   s_row[16];
  __shared__ int   s_col[16];

  const int lane = threadIdx.x;
  const int e0 = blockIdx.x * 16;
  const int hh = lane >> 4, nn = lane & 15;

  if (lane < 16) {
    const int e = e0 + lane;
    const int r = ei[e], c = ei[E + e];
    s_row[lane] = r;
    s_col[lane] = c;
    float dx = coord[3 * r + 0] - coord[3 * c + 0];
    float dy = coord[3 * r + 1] - coord[3 * c + 1];
    float dz = coord[3 * r + 2] - coord[3 * c + 2];
    s_diff[lane][0] = dx; s_diff[lane][1] = dy; s_diff[lane][2] = dz;
    s_ef[frag_off(lane, 2 * NIN)]     = (_Float16)(dx * dx + dy * dy + dz * dz);
    s_ef[frag_off(lane, 2 * NIN + 1)] = (_Float16)eattr[e];
#pragma unroll
    for (int k = 2 * NIN + 2; k < K1P; ++k) s_ef[frag_off(lane, k)] = (_Float16)0.f;
  }
  if (NHK > NH) {  // zero the K-pad columns used by GEMM2/GEMM3
    for (int idx = lane; idx < 16 * (NHK - NH); idx += 32) {
      int m = idx / (NHK - NH), k = NH + idx % (NHK - NH);
      s_m1[frag_off(m, k)] = (_Float16)0.f;
      s_m2[frag_off(m, k)] = (_Float16)0.f;
    }
  }
  __syncthreads();

  for (int idx = lane; idx < 16 * NIN; idx += 32) {
    const int m = idx / NIN, f = idx % NIN;
    s_ef[frag_off(m, f)]       = (_Float16)h[s_row[m] * NIN + f];
    s_ef[frag_off(m, NIN + f)] = (_Float16)h[s_col[m] * NIN + f];
  }
  __syncthreads();

  // GEMM1: m1 = silu(ef @ We1 + be1)
#pragma unroll
  for (int nt = 0; nt < NH / 16; ++nt) {
    v8f acc = {};
#pragma unroll
    for (int kt = 0; kt < K1P / 32; ++kt)
      acc = wmma16(load_frag(s_ef, lane, kt),
                   load_fragW(We1, lane, kt, nt, K1P / 32), acc);
    const int n = nt * 16 + nn;
    const float bv = be1[n];
#pragma unroll
    for (int j = 0; j < 8; ++j)
      s_m1[frag_off(j + 8 * hh, n)] = (_Float16)siluf(acc[j] + bv);
  }
  __syncthreads();

  // GEMM2: m2 = silu(m1 @ We2 + be2); scatter segment-sum into agg
#pragma unroll
  for (int nt = 0; nt < NH / 16; ++nt) {
    v8f acc = {};
#pragma unroll
    for (int kt = 0; kt < NHK / 32; ++kt)
      acc = wmma16(load_frag(s_m1, lane, kt),
                   load_fragW(We2, lane, kt, nt, NHK / 32), acc);
    const int n = nt * 16 + nn;
    const float bv = be2[n];
#pragma unroll
    for (int j = 0; j < 8; ++j) {
      const int mr = j + 8 * hh;
      const float v = siluf(acc[j] + bv);
      s_m2[frag_off(mr, n)] = (_Float16)v;
      atomicAdd(&agg[s_row[mr] * NH + n], v);
    }
  }
  __syncthreads();

  // GEMM3: t = silu(m2 @ Wc1 + bc1)
#pragma unroll
  for (int nt = 0; nt < NH / 16; ++nt) {
    v8f acc = {};
#pragma unroll
    for (int kt = 0; kt < NHK / 32; ++kt)
      acc = wmma16(load_frag(s_m2, lane, kt),
                   load_fragW(Wc1, lane, kt, nt, NHK / 32), acc);
    const int n = nt * 16 + nn;
    const float bv = bc1[n];
#pragma unroll
    for (int j = 0; j < 8; ++j) s_t[j + 8 * hh][n] = siluf(acc[j] + bv);
  }
  __syncthreads();

  // ct = t . Wc2 ; coord-delta scatter
  if (lane < 16) {
    float ct = 0.f;
#pragma unroll
    for (int k = 0; k < NH; ++k) ct += s_t[lane][k] * Wc2[k];
    const int r = s_row[lane];
    atomicAdd(&cdelta[3 * r + 0], s_diff[lane][0] * ct);
    atomicAdd(&cdelta[3 * r + 1], s_diff[lane][1] * ct);
    atomicAdd(&cdelta[3 * r + 2], s_diff[lane][2] * ct);
  }
}

// ---------------------------------------------------------------------------
// EGCL node update: h2 = silu([h,agg]@Wn1+bn1)@Wn2 + bn2
// ---------------------------------------------------------------------------
template <int NIN, int NH, int NOUT>
__global__ __launch_bounds__(32) void k_node(
    const float* __restrict__ h, const float* __restrict__ agg,
    const _Float16* __restrict__ Wn1, const float* __restrict__ bn1,
    const _Float16* __restrict__ Wn2, const float* __restrict__ bn2,
    float* __restrict__ out, int N) {
  constexpr int K2  = NIN + NH;      // 32 / 64 / 128
  constexpr int NHK = (NH < 32) ? 32 : NH;
  __shared__ alignas(32) _Float16 s_a[16 * K2];
  __shared__ alignas(32) _Float16 s_hid[16 * NHK];
  const int lane = threadIdx.x;
  const int n0 = blockIdx.x * 16;
  const int hh = lane >> 4, nn = lane & 15;

  for (int idx = lane; idx < 16 * NIN; idx += 32) {
    const int m = idx / NIN, f = idx % NIN;
    s_a[frag_off(m, f)] = (_Float16)h[(n0 + m) * NIN + f];
  }
  for (int idx = lane; idx < 16 * NH; idx += 32) {
    const int m = idx / NH, f = idx % NH;
    s_a[frag_off(m, NIN + f)] = (_Float16)agg[(n0 + m) * NH + f];
  }
  if (NHK > NH) {
    for (int idx = lane; idx < 16 * (NHK - NH); idx += 32) {
      int m = idx / (NHK - NH), k = NH + idx % (NHK - NH);
      s_hid[frag_off(m, k)] = (_Float16)0.f;
    }
  }
  __syncthreads();

#pragma unroll
  for (int nt = 0; nt < NH / 16; ++nt) {
    v8f acc = {};
#pragma unroll
    for (int kt = 0; kt < K2 / 32; ++kt)
      acc = wmma16(load_frag(s_a, lane, kt),
                   load_fragW(Wn1, lane, kt, nt, K2 / 32), acc);
    const int n = nt * 16 + nn;
    const float bv = bn1[n];
#pragma unroll
    for (int j = 0; j < 8; ++j)
      s_hid[frag_off(j + 8 * hh, n)] = (_Float16)siluf(acc[j] + bv);
  }
  __syncthreads();

#pragma unroll
  for (int nt = 0; nt < NOUT / 16; ++nt) {
    v8f acc = {};
#pragma unroll
    for (int kt = 0; kt < NHK / 32; ++kt)
      acc = wmma16(load_frag(s_hid, lane, kt),
                   load_fragW(Wn2, lane, kt, nt, NHK / 32), acc);
    const int n = nt * 16 + nn;
    const float bv = bn2[n];
#pragma unroll
    for (int j = 0; j < 8; ++j)
      out[(n0 + j + 8 * hh) * NOUT + n] = acc[j] + bv;
  }
}

// ---------------------------------------------------------------------------
// lin1: relu(x @ W + b), K=NOUT=128
// ---------------------------------------------------------------------------
template <int K, int NOUT>
__global__ __launch_bounds__(32) void k_dense_relu(
    const float* __restrict__ in, const _Float16* __restrict__ W,
    const float* __restrict__ b, float* __restrict__ out, int N) {
  __shared__ alignas(32) _Float16 s_a[16 * K];
  const int lane = threadIdx.x;
  const int n0 = blockIdx.x * 16;
  const int hh = lane >> 4, nn = lane & 15;
  for (int idx = lane; idx < 16 * K; idx += 32) {
    const int m = idx / K, f = idx % K;
    s_a[frag_off(m, f)] = (_Float16)in[(n0 + m) * K + f];
  }
  __syncthreads();
#pragma unroll
  for (int nt = 0; nt < NOUT / 16; ++nt) {
    v8f acc = {};
#pragma unroll
    for (int kt = 0; kt < K / 32; ++kt)
      acc = wmma16(load_frag(s_a, lane, kt),
                   load_fragW(W, lane, kt, nt, K / 32), acc);
    const int n = nt * 16 + nn;
    const float bv = b[n];
#pragma unroll
    for (int j = 0; j < 8; ++j)
      out[(n0 + j + 8 * hh) * NOUT + n] = fmaxf(acc[j] + bv, 0.f);
  }
}

// ---------------------------------------------------------------------------
// Global mean pool + final linear + log_softmax (G=8, TD=10 -> trivial)
// ---------------------------------------------------------------------------
__global__ void k_pool(const float* __restrict__ x, const int* __restrict__ batch,
                       float* gsum, int N) {
  int idx = blockIdx.x * blockDim.x + threadIdx.x;
  if (idx >= N * 128) return;
  int i = idx >> 7, f = idx & 127;
  atomicAdd(&gsum[batch[i] * 128 + f], x[idx]);
}

__global__ void k_head(const float* __restrict__ gsum, const float* __restrict__ ccnt,
                       const float* __restrict__ W, const float* __restrict__ b,
                       float* __restrict__ out) {
  __shared__ float sm[8][10];
  int t = threadIdx.x;
  if (t < 80) {
    int g = t / 10, o = t % 10;
    float inv = 1.f / fmaxf(ccnt[g], 1.f);
    float acc = b[o];
    for (int k = 0; k < 128; ++k) acc += gsum[g * 128 + k] * inv * W[k * 10 + o];
    sm[g][o] = acc;
  }
  __syncthreads();
  if (t < 8) {
    float mx = -3.4e38f;
    for (int o = 0; o < 10; ++o) mx = fmaxf(mx, sm[t][o]);
    float s = 0.f;
    for (int o = 0; o < 10; ++o) s += __expf(sm[t][o] - mx);
    float lse = mx + __logf(s);
    for (int o = 0; o < 10; ++o) out[t * 10 + o] = sm[t][o] - lse;
  }
}

// ---------------------------------------------------------------------------
// Host orchestration
// ---------------------------------------------------------------------------
extern "C" void kernel_launch(void* const* d_in, const int* in_sizes, int n_in,
                              void* d_out, int out_size, void* d_ws, size_t ws_size,
                              hipStream_t stream) {
  (void)in_sizes; (void)out_size;
  if (n_in < 46) return;
  constexpr int N = 100000, E = 1200000, F = 200000, V2F = 600000;

  const float* pos   = (const float*)d_in[0];
  const int*   batch = (const int*)d_in[1];
  const int*   face  = (const int*)d_in[2];
  const int*   ei    = (const int*)d_in[3];
  const float* eattr = (const float*)d_in[4];
  const int*   v2fv  = (const int*)d_in[5];
  const int*   v2ff  = (const int*)d_in[6];
  const float* featW = (const float*)d_in[7];
  const float* featb = (const float*)d_in[8];
  const float* lin1b = (const float*)d_in[43];
  const float* lin2W = (const float*)d_in[44];
  const float* lin2b = (const float*)d_in[45];

  // per-conv fp32 bias / Wc2 pointers (base: conv1=9, conv2=20, conv3=31)
  const int cb[3] = {9, 20, 31};
  const float *be1[3], *be2[3], *bn1[3], *bn2[3], *bc1[3], *Wc2[3];
  for (int l = 0; l < 3; ++l) {
    be1[l] = (const float*)d_in[cb[l] + 1];
    be2[l] = (const float*)d_in[cb[l] + 3];
    bn1[l] = (const float*)d_in[cb[l] + 5];
    bn2[l] = (const float*)d_in[cb[l] + 7];
    bc1[l] = (const float*)d_in[cb[l] + 9];
    Wc2[l] = (const float*)d_in[cb[l] + 10];
  }

  // workspace carve-out
  char* ws = (char*)d_ws;
  size_t off = 0;
  auto alloc = [&](size_t bytes) -> void* {
    void* p = ws + off;
    off += (bytes + 255) & ~size_t(255);
    return p;
  };
  float* coordA = (float*)alloc((size_t)N * 3 * 4);
  float* coordB = (float*)alloc((size_t)N * 3 * 4);
  float* hA     = (float*)alloc((size_t)N * 128 * 4);
  float* hB     = (float*)alloc((size_t)N * 128 * 4);
  float* agg    = (float*)alloc((size_t)N * 64 * 4);
  float* cdelta = (float*)alloc((size_t)N * 3 * 4);
  float* cntrow = (float*)alloc((size_t)N * 4);
  float* asum   = (float*)alloc((size_t)N * 4);
  float* acnt   = (float*)alloc((size_t)N * 4);
  float* farea  = (float*)alloc((size_t)F * 4);
  float* small  = (float*)alloc(1064 * 4);
  float* csum = small;                              // 24
  float* ccnt = small + 24;                         // 8
  unsigned int* mrad = (unsigned int*)(small + 32); // 8
  float* gsum = small + 40;                         // 1024
  _Float16* wbase = (_Float16*)alloc(64000 * sizeof(_Float16));
  if (off > ws_size) return;

  // f16 weight repack specs (src idx, K, N, Kpad, Npad), order: per conv
  // We1, We2, Wn1, Wn2, Wc1 ; then lin1.
  struct Spec { int src, K, Nn, KP, NP; };
  const Spec sp[16] = {
      {9, 34, 16, 64, 16},   {11, 16, 16, 32, 16},  {13, 32, 16, 32, 16},
      {15, 16, 32, 32, 32},  {17, 16, 16, 32, 16},
      {20, 66, 32, 96, 32},  {22, 32, 32, 32, 32},  {24, 64, 32, 64, 32},
      {26, 32, 64, 32, 64},  {28, 32, 32, 32, 32},
      {31, 130, 64, 160, 64},{33, 64, 64, 64, 64},  {35, 128, 64, 128, 64},
      {37, 64, 128, 64, 128},{39, 64, 64, 64, 64},
      {42, 128, 128, 128, 128}};
  _Float16* wp[16];
  {
    size_t woff = 0;
    for (int i = 0; i < 16; ++i) {
      wp[i] = wbase + woff;
      woff += (size_t)sp[i].KP * sp[i].NP;
    }
  }
  for (int i = 0; i < 16; ++i) {
    int tot = sp[i].KP * sp[i].NP;
    k_cvt<<<(tot + 255) / 256, 256, 0, stream>>>(
        (const float*)d_in[sp[i].src], wp[i], sp[i].K, sp[i].Nn, sp[i].KP, sp[i].NP);
  }

  // zero accumulators
  hipMemsetAsync(small, 0, 1064 * 4, stream);
  hipMemsetAsync(cntrow, 0, (size_t)N * 4, stream);
  hipMemsetAsync(asum, 0, (size_t)N * 4, stream);
  hipMemsetAsync(acnt, 0, (size_t)N * 4, stream);

  const int nbN = (N + 255) / 256;
  // pos normalization
  k_centroid<<<nbN, 256, 0, stream>>>(pos, batch, csum, ccnt, N);
  k_norm1<<<nbN, 256, 0, stream>>>(pos, batch, csum, ccnt, coordA, mrad, N);
  k_norm2<<<nbN, 256, 0, stream>>>(coordA, batch, mrad, N);
  // face area -> vertex area -> node features
  k_farea<<<(F + 255) / 256, 256, 0, stream>>>(coordA, face, farea, F);
  k_v2f<<<(V2F + 255) / 256, 256, 0, stream>>>(v2fv, v2ff, farea, asum, acnt, V2F);
  k_feat<<<nbN, 256, 0, stream>>>(asum, acnt, featW, featb, hA, N);
  // edge row counts (constant across layers)
  k_cntrow<<<(E + 255) / 256, 256, 0, stream>>>(ei, cntrow, E);

  // ---- EGCL layer 1: h(16)@coordA -> h(32)@coordB ----
  hipMemsetAsync(agg, 0, (size_t)N * 16 * 4, stream);
  hipMemsetAsync(cdelta, 0, (size_t)N * 3 * 4, stream);
  k_edge<16, 16><<<E / 16, 32, 0, stream>>>(ei, eattr, hA, coordA,
      wp[0], be1[0], wp[1], be2[0], wp[4], bc1[0], Wc2[0], agg, cdelta, E);
  k_coordupd<<<nbN, 256, 0, stream>>>(coordA, cdelta, cntrow, coordB, N);
  k_node<16, 16, 32><<<N / 16, 32, 0, stream>>>(hA, agg, wp[2], bn1[0],
                                                wp[3], bn2[0], hB, N);

  // ---- EGCL layer 2: h(32)@coordB -> h(64)@coordA ----
  hipMemsetAsync(agg, 0, (size_t)N * 32 * 4, stream);
  hipMemsetAsync(cdelta, 0, (size_t)N * 3 * 4, stream);
  k_edge<32, 32><<<E / 16, 32, 0, stream>>>(ei, eattr, hB, coordB,
      wp[5], be1[1], wp[6], be2[1], wp[9], bc1[1], Wc2[1], agg, cdelta, E);
  k_coordupd<<<nbN, 256, 0, stream>>>(coordB, cdelta, cntrow, coordA, N);
  k_node<32, 32, 64><<<N / 16, 32, 0, stream>>>(hB, agg, wp[7], bn1[1],
                                                wp[8], bn2[1], hA, N);

  // ---- EGCL layer 3: h(64)@coordA -> h(128)@coordB ----
  hipMemsetAsync(agg, 0, (size_t)N * 64 * 4, stream);
  hipMemsetAsync(cdelta, 0, (size_t)N * 3 * 4, stream);
  k_edge<64, 64><<<E / 16, 32, 0, stream>>>(ei, eattr, hA, coordA,
      wp[10], be1[2], wp[11], be2[2], wp[14], bc1[2], Wc2[2], agg, cdelta, E);
  k_coordupd<<<nbN, 256, 0, stream>>>(coordA, cdelta, cntrow, coordB, N);
  k_node<64, 64, 128><<<N / 16, 32, 0, stream>>>(hA, agg, wp[12], bn1[2],
                                                 wp[13], bn2[2], hB, N);

  // ---- head: relu(lin1) -> mean pool -> lin2 -> log_softmax ----
  k_dense_relu<128, 128><<<N / 16, 32, 0, stream>>>(hB, wp[15], lin1b, hA, N);
  k_pool<<<((size_t)N * 128 + 255) / 256, 256, 0, stream>>>(hA, batch, gsum, N);
  k_head<<<1, 128, 0, stream>>>(gsum, ccnt, lin2W, lin2b, (float*)d_out);
}
